// minGRU_14894946583333
// MI455X (gfx1250) — matrix operand order repaired and checked by reading, not compiled, not run on hardware
//
#include <hip/hip_runtime.h>
#include <hip/hip_bf16.h>
#include <math.h>

// ---------------------------------------------------------------------------
// Problem constants (from reference): B=4, L=4096, D=1024
// ---------------------------------------------------------------------------
#define BB 4
#define LL 4096
#define DD 1024
#define MM (BB * LL)      // 16384 GEMM rows
#define NN (2 * DD)       // 2048  GEMM cols (hidden | gate)
#define KK DD             // 1024  reduction
#define NCHUNK 64         // scan chunks along L
#define CLEN   64         // chunk length (NCHUNK*CLEN == LL)

typedef __attribute__((ext_vector_type(16))) __bf16 v16bf;
typedef __attribute__((ext_vector_type(8)))  __bf16 v8bf;
typedef __attribute__((ext_vector_type(8)))  float  v8f;
typedef __attribute__((ext_vector_type(4)))  float  f32x4;

// Fragment assembled from two 16-byte global_load_b128's, no VALU packing.
union Frag16 {
    v16bf v;
    f32x4 q[2];
};

// ---------------------------------------------------------------------------
// f32 -> bf16 pre-conversion (8 elements / thread: 2x b128 load, 1x b128 store)
// ---------------------------------------------------------------------------
__global__ __launch_bounds__(256)
void cvt_f32_to_bf16(const float* __restrict__ in, __bf16* __restrict__ out) {
    const size_t i = ((size_t)blockIdx.x * blockDim.x + threadIdx.x) * 8;
    f32x4 a = *(const f32x4*)(in + i);
    f32x4 b = *(const f32x4*)(in + i + 4);
    v8bf o;
#pragma unroll
    for (int j = 0; j < 4; ++j) { o[j] = (__bf16)a[j]; o[j + 4] = (__bf16)b[j]; }
    *(v8bf*)(out + i) = o;
}

// ---------------------------------------------------------------------------
// GEMM: hg[m][n] = sum_k x[m][k] * W[n][k]   (bf16 inputs, f32 accumulate)
// Per-wave 64x64 tile = 4x4 WMMA subtiles; 4 waves/WG arranged 2x2 -> 128x128.
// A fragment (16-bit 16x32): lane l (l<16): row M=l, elems 0..7 = K 0..7,
//   elems 8..15 = K 16..23 ; lanes 16..31: same rows, K 8..15 / 24..31.
// B fragment (16-bit 32x16): lane l: col N=l%16, same K order as A
//   (elems 0..7 = K kb..kb+7, 8..15 = K kb+16..kb+23, kb = (l<16 ? 0 : 8)).  Both are contiguous bf16 runs in memory,
//   so fragments load directly via global_load_b128 (zero conversion VALU).
// C/D (f32 16x16): vgpr r, lanes 0-15: (M=r, N=lane); lanes 16-31: (M=r+8).
// ---------------------------------------------------------------------------
__global__ __launch_bounds__(128)
void gemm_hg_bf16(const __bf16* __restrict__ X, const __bf16* __restrict__ W,
                  float* __restrict__ HG) {
    const int lane  = threadIdx.x & 31;
    const int wave  = threadIdx.x >> 5;
    const int wm    = wave >> 1;               // 0..1
    const int wn    = wave & 1;                // 0..1
    const int m0    = blockIdx.x * 128 + wm * 64;
    const int n0    = blockIdx.y * 128 + wn * 64;
    const int lh    = lane & 15;
    const int khalf = lane >> 4;

    v8f acc[4][4];
#pragma unroll
    for (int i = 0; i < 4; ++i)
#pragma unroll
        for (int j = 0; j < 4; ++j)
#pragma unroll
            for (int r = 0; r < 8; ++r) acc[i][j][r] = 0.0f;

    // Per-lane base pointers (K offsets advance by 32 each iteration)
    const __bf16* ap[4];
    const __bf16* bp[4];
#pragma unroll
    for (int i = 0; i < 4; ++i)
        ap[i] = X + (size_t)(m0 + i * 16 + lh) * KK + khalf * 8;
#pragma unroll
    for (int j = 0; j < 4; ++j)
        bp[j] = W + (size_t)(n0 + j * 16 + lh) * KK + khalf * 8;

    for (int k = 0; k < KK; k += 32) {
        Frag16 a[4], b[4];
#pragma unroll
        for (int i = 0; i < 4; ++i) {
            a[i].q[0] = *(const f32x4*)(ap[i] + k);        // K run [kb, kb+8)
            a[i].q[1] = *(const f32x4*)(ap[i] + k + 16);   // K run [kb+16, kb+24)
        }
#pragma unroll
        for (int j = 0; j < 4; ++j) {
            b[j].q[0] = *(const f32x4*)(bp[j] + k);        // K run [kb, kb+8)
            b[j].q[1] = *(const f32x4*)(bp[j] + k + 16);   // K run [kb+16, kb+24)
        }
#pragma unroll
        for (int i = 0; i < 4; ++i)
#pragma unroll
            for (int j = 0; j < 4; ++j)
                acc[i][j] = __builtin_amdgcn_wmma_f32_16x16x32_bf16(
                    false, a[i].v, false, b[j].v, (short)0, acc[i][j], false, false);
    }

    // Store: vgpr r of tile (i,j): row m0+i*16+r+(lane>=16?8:0), col n0+j*16+(lane&15)
#pragma unroll
    for (int i = 0; i < 4; ++i) {
        const int mbase = m0 + i * 16 + khalf * 8;
#pragma unroll
        for (int j = 0; j < 4; ++j) {
            const int n = n0 + j * 16 + lh;
#pragma unroll
            for (int r = 0; r < 8; ++r)
                HG[(size_t)(mbase + r) * NN + n] = acc[i][j][r];
        }
    }
}

// ---------------------------------------------------------------------------
// Elementwise log-space helpers
// ---------------------------------------------------------------------------
__device__ __forceinline__ float softplusf(float x) {
    // softplus(x) = max(x,0) + log1p(exp(-|x|))
    float r = __logf(1.0f + __expf(-fabsf(x)));
    return (x > 0.0f) ? x + r : r;
}
__device__ __forceinline__ float log_gf(float x) {
    return (x >= 0.0f) ? __logf(x + 0.5f) : -softplusf(-x);
}
__device__ __forceinline__ void lse_acc(float v, float& m, float& s) {
    if (v > m) { s = s * __expf(m - v) + 1.0f; m = v; }
    else       { s += __expf(v - m); }
}

// ---------------------------------------------------------------------------
// Pass 1: per (b, chunk, d) compute chunk-local  A = sum(log_coeffs)  and
//         chunkL = logsumexp over chunk of (log_values - local_cumsum)
// grid: (D/256, NCHUNK, B), block 256 -> loads coalesced across d.
// ---------------------------------------------------------------------------
__global__ __launch_bounds__(256)
void scan_partial(const float* __restrict__ HG, float* __restrict__ chunkA,
                  float* __restrict__ chunkL) {
    const int d     = blockIdx.x * blockDim.x + threadIdx.x;
    const int chunk = blockIdx.y;
    const int b     = blockIdx.z;
    const size_t base = (size_t)b * LL * NN;

    float asum = 0.0f, m = -1e30f, s = 0.0f;
    const int l0 = chunk * CLEN;
    for (int i = 0; i < CLEN; ++i) {
        const size_t row = base + (size_t)(l0 + i) * NN;
        float h = HG[row + d];
        float g = HG[row + DD + d];
        float lc = -softplusf(g);
        float lv = -softplusf(-g) + log_gf(h);
        asum += lc;
        lse_acc(lv - asum, m, s);
    }
    const size_t idx = ((size_t)b * NCHUNK + chunk) * DD + d;
    chunkA[idx] = asum;
    chunkL[idx] = m + __logf(s);
}

// ---------------------------------------------------------------------------
// Pass 2: per channel, exclusive scan over the 64 chunk totals.
// Overwrites chunkA[idx] with Aprev(chunk) and writes carryC(chunk) =
// logsumexp over all earlier chunks of (log_values - a_star)  (-3e38 = "-inf").
// ---------------------------------------------------------------------------
__global__ __launch_bounds__(256)
void scan_chunks(float* __restrict__ chunkA, const float* __restrict__ chunkL,
                 float* __restrict__ carryC) {
    const int t = blockIdx.x * blockDim.x + threadIdx.x;   // 0 .. B*D-1
    const int b = t / DD;
    const int d = t % DD;

    float Aprev = 0.0f, m = -1e30f, s = 0.0f;
    for (int c = 0; c < NCHUNK; ++c) {
        const size_t idx = ((size_t)b * NCHUNK + c) * DD + d;
        float a   = chunkA[idx];
        float lco = chunkL[idx];
        chunkA[idx] = Aprev;                                  // exclusive prefix
        carryC[idx] = (s > 0.0f) ? (m + __logf(s)) : -3.0e38f;
        lse_acc(lco - Aprev, m, s);                           // fold chunk total
        Aprev += a;
    }
}

// ---------------------------------------------------------------------------
// Pass 3: replay each chunk with its carries; out = exp(a_star + cumlogsumexp)
// ---------------------------------------------------------------------------
__global__ __launch_bounds__(256)
void scan_final(const float* __restrict__ HG, const float* __restrict__ AprevArr,
                const float* __restrict__ carryC, float* __restrict__ out) {
    const int d     = blockIdx.x * blockDim.x + threadIdx.x;
    const int chunk = blockIdx.y;
    const int b     = blockIdx.z;
    const size_t cidx = ((size_t)b * NCHUNK + chunk) * DD + d;

    const float Aprev = AprevArr[cidx];
    const float C     = carryC[cidx];
    float m = -1e30f, s = 0.0f;
    if (C > -1.0e37f) { m = C; s = 1.0f; }

    float aloc = 0.0f;
    const int l0 = chunk * CLEN;
    const size_t base = (size_t)b * LL * NN;
    for (int i = 0; i < CLEN; ++i) {
        const int l = l0 + i;
        const size_t row = base + (size_t)l * NN;
        float h = HG[row + d];
        float g = HG[row + DD + d];
        float lc = -softplusf(g);
        float lv = -softplusf(-g) + log_gf(h);
        aloc += lc;
        lse_acc(lv - aloc - Aprev, m, s);
        out[((size_t)b * LL + l) * DD + d] = __expf(Aprev + aloc + m) * s;
    }
}

// ---------------------------------------------------------------------------
// Launch. Workspace layout (~167 MB):
//   hg:      M*N floats            = 128 MB
//   chunkA/chunkL/carryC:          =   3 MB
//   Xb:      M*K bf16              =  32 MB
//   Wb:      N*K bf16              =   4 MB
// ---------------------------------------------------------------------------
extern "C" void kernel_launch(void* const* d_in, const int* in_sizes, int n_in,
                              void* d_out, int out_size, void* d_ws, size_t ws_size,
                              hipStream_t stream) {
    (void)in_sizes; (void)n_in; (void)out_size; (void)ws_size;
    const float* x = (const float*)d_in[0];
    const float* W = (const float*)d_in[1];
    float* out = (float*)d_out;

    float*  hg     = (float*)d_ws;
    float*  chunkA = hg + (size_t)MM * NN;
    float*  chunkL = chunkA + (size_t)BB * NCHUNK * DD;
    float*  carryC = chunkL + (size_t)BB * NCHUNK * DD;
    __bf16* Xb     = (__bf16*)(carryC + (size_t)BB * NCHUNK * DD);
    __bf16* Wb     = Xb + (size_t)MM * KK;

    // Pre-convert inputs to bf16 (8 elems/thread)
    cvt_f32_to_bf16<<<(size_t)MM * KK / (8 * 256), 256, 0, stream>>>(x, Xb);
    cvt_f32_to_bf16<<<(size_t)NN * KK / (8 * 256), 256, 0, stream>>>(W, Wb);

    dim3 gGemm(MM / 128, NN / 128);                // 128 x 16 workgroups
    gemm_hg_bf16<<<gGemm, 128, 0, stream>>>(Xb, Wb, hg);

    dim3 gScan(DD / 256, NCHUNK, BB);              // 4 x 64 x 4
    scan_partial<<<gScan, 256, 0, stream>>>(hg, chunkA, chunkL);

    scan_chunks<<<(BB * DD) / 256, 256, 0, stream>>>(chunkA, chunkL, carryC);

    scan_final<<<gScan, 256, 0, stream>>>(hg, chunkA, carryC, out);
}
